// MultiAttention_45835890983057
// MI455X (gfx1250) — compile-verified, matrix operand order checked by reading, not run on hardware
//
#include <hip/hip_runtime.h>
#include <stdint.h>

// CDNA5 / gfx1250 — wave32, WMMA 16x16x32 bf16, f32 accumulate.

typedef float  v8f   __attribute__((ext_vector_type(8)));
typedef __bf16 v16bf __attribute__((ext_vector_type(16)));

union BfVec { v16bf v; uint32_t u[8]; };

__device__ __forceinline__ uint32_t pack_bf16(float lo, float hi) {
    union { __bf16 h; uint16_t u; } a, b;
    a.h = (__bf16)lo; b.h = (__bf16)hi;
    return (uint32_t)a.u | ((uint32_t)b.u << 16);
}

// K-offset of the bf16 pair held in "vgpr" j for A/B operands (ISA 7.12.2):
// VGPR 0..3 -> K = 2j   (+8 for lanes 16..31)
// VGPR 4..7 -> K = 2j+8 (+8 for lanes 16..31)
__device__ __forceinline__ int koff(int j, int kh) {
    return 2 * j + ((j >= 4) ? 8 : 0) + kh;
}

#define SQ   2048
#define DMODEL 1024
#define HEADS 16
#define KEYD 64

// ---------------------------------------------------------------------------
// Projection: Y = X @ W, X:(B*S,1024) f32, W:(1024,1024) f32.
// Output bf16, layout [bh][s][64] (transpose_out=0) or [bh][64][s] (=1, for V).
// One wave per 16(M)x64(N) tile; K loop 1024 in steps of 32; 4 WMMA per step.
// ---------------------------------------------------------------------------
__global__ __launch_bounds__(256)
void proj_kernel(const float* __restrict__ X, const float* __restrict__ W,
                 __bf16* __restrict__ Y, int transpose_out) {
    const int lane = threadIdx.x & 31;
    const int wid  = threadIdx.x >> 5;
    const int wg   = blockIdx.x * 8 + wid;       // 0..4095
    const int tile_m = wg >> 4;                  // 0..255
    const int h      = wg & 15;                  // head / 64-col group
    const int m0   = tile_m * 16;                // row in (B*S)
    const int kh   = (lane >> 4) * 8;
    const int lid  = lane & 15;                  // A row / B col within tile
    const int half = lane >> 4;

    v8f acc[4] = {};
    for (int k0 = 0; k0 < DMODEL; k0 += 32) {
        BfVec A;
        const float* xr = X + (size_t)(m0 + lid) * DMODEL + k0;
        #pragma unroll
        for (int j = 0; j < 8; ++j) {
            int kk = koff(j, kh);
            A.u[j] = pack_bf16(xr[kk], xr[kk + 1]);
        }
        #pragma unroll
        for (int t = 0; t < 4; ++t) {
            BfVec Bm;
            const float* wc = W + (size_t)k0 * DMODEL + h * KEYD + t * 16 + lid;
            #pragma unroll
            for (int j = 0; j < 8; ++j) {
                int kk = koff(j, kh);
                Bm.u[j] = pack_bf16(wc[(size_t)kk * DMODEL],
                                    wc[(size_t)(kk + 1) * DMODEL]);
            }
            acc[t] = __builtin_amdgcn_wmma_f32_16x16x32_bf16(
                false, A.v, false, Bm.v, (short)0, acc[t], false, false);
        }
    }

    const int b  = m0 >> 11;            // row / 2048
    const int s0 = m0 & (SQ - 1);
    #pragma unroll
    for (int t = 0; t < 4; ++t) {
        #pragma unroll
        for (int r = 0; r < 8; ++r) {
            int m = r + 8 * half;                // C/D: vgpr r, half -> row
            int srow = s0 + m;
            int d = t * 16 + lid;                // C/D: lane -> col
            size_t idx;
            if (transpose_out)
                idx = (((size_t)(b * HEADS + h) * KEYD) + d) * SQ + srow;
            else
                idx = (((size_t)(b * HEADS + h) * SQ) + srow) * KEYD + d;
            Y[idx] = (__bf16)acc[t][r];
        }
    }
}

// ---------------------------------------------------------------------------
// Fused causal attention, flash-style online softmax.
// One wave per 16-row q-tile; 32 kv columns per step:
//   4 WMMA for scores (K=64 chained 2x32), softmax stats via 16-lane shfl_xor,
//   P -> bf16 through LDS (D-layout -> A-layout), 4 WMMA for PV.
// ---------------------------------------------------------------------------
__global__ __launch_bounds__(256)
void attn_kernel(const __bf16* __restrict__ Qw, const __bf16* __restrict__ Kw,
                 const __bf16* __restrict__ VwT, float* __restrict__ Out) {
    __shared__ uint16_t Plds[8][16 * 32];

    const int lane = threadIdx.x & 31;
    const int wid  = threadIdx.x >> 5;
    const int wg   = blockIdx.x * 8 + wid;       // 0..4095
    const int bh   = wg >> 7;                    // b*16 + h
    const int q0   = (wg & 127) * 16;
    const int kh   = (lane >> 4) * 8;
    const int half = lane >> 4;
    const int lid  = lane & 15;

    // Q tile A-operands (16x64 = two 16x32 slabs), loaded once
    BfVec Aq[2];
    const __bf16* qrow = Qw + ((size_t)bh * SQ + q0 + lid) * KEYD;
    #pragma unroll
    for (int c = 0; c < 2; ++c)
        #pragma unroll
        for (int j = 0; j < 8; ++j)
            Aq[c].u[j] = *(const uint32_t*)(qrow + c * 32 + koff(j, kh));

    v8f acc[4] = {};
    float mrow[8], lrow[8];
    #pragma unroll
    for (int r = 0; r < 8; ++r) { mrow[r] = -1e30f; lrow[r] = 0.f; }

    const int kv_end = q0 + 16;                  // causal upper bound (excl)
    for (int kv0 = 0; kv0 < kv_end; kv0 += 32) {
        // ---- scores: two 16-col tiles, K-dim 64 chained via C operand ----
        v8f s[2];
        #pragma unroll
        for (int g = 0; g < 2; ++g) {
            v8f sc = {};
            #pragma unroll
            for (int c = 0; c < 2; ++c) {
                BfVec Bk;   // B[d][kv]: lane -> kv col, vgpr -> d pair (contig)
                const __bf16* krow =
                    Kw + ((size_t)bh * SQ + kv0 + g * 16 + lid) * KEYD + c * 32;
                #pragma unroll
                for (int j = 0; j < 8; ++j)
                    Bk.u[j] = *(const uint32_t*)(krow + koff(j, kh));
                sc = __builtin_amdgcn_wmma_f32_16x16x32_bf16(
                    false, Aq[c].v, false, Bk.v, (short)0, sc, false, false);
            }
            s[g] = sc;
        }

        // ---- online softmax over these 32 columns ----
        float p0[8], p1[8], rescale[8];
        #pragma unroll
        for (int r = 0; r < 8; ++r) {
            int m  = r + 8 * half;
            int qg = q0 + m;
            float v0 = s[0][r] * 0.125f;         // 1/sqrt(64)
            float v1 = s[1][r] * 0.125f;
            if (kv0 + lid > qg)      v0 = -1e30f;  // causal mask
            if (kv0 + 16 + lid > qg) v1 = -1e30f;
            float mx = fmaxf(v0, v1);
            #pragma unroll
            for (int o = 1; o < 16; o <<= 1)
                mx = fmaxf(mx, __shfl_xor(mx, o, 16));
            float mn = fmaxf(mrow[r], mx);
            rescale[r] = __expf(mrow[r] - mn);
            mrow[r] = mn;
            p0[r] = __expf(v0 - mn);
            p1[r] = __expf(v1 - mn);
            float sum = p0[r] + p1[r];
            #pragma unroll
            for (int o = 1; o < 16; o <<= 1)
                sum += __shfl_xor(sum, o, 16);
            lrow[r] = lrow[r] * rescale[r] + sum;
        }
        #pragma unroll
        for (int t = 0; t < 4; ++t)
            #pragma unroll
            for (int r = 0; r < 8; ++r)
                acc[t][r] *= rescale[r];

        // ---- P: C/D-layout f32 -> A-layout bf16 via per-wave LDS tile ----
        #pragma unroll
        for (int r = 0; r < 8; ++r) {
            int m = r + 8 * half;
            union { __bf16 h; uint16_t u; } cv;
            cv.h = (__bf16)p0[r]; Plds[wid][m * 32 + lid]      = cv.u;
            cv.h = (__bf16)p1[r]; Plds[wid][m * 32 + 16 + lid] = cv.u;
        }
        // same-wave LDS ops are in-order; wait + clobber pins compiler order
        asm volatile("s_wait_dscnt 0" ::: "memory");

        BfVec Ap;
        #pragma unroll
        for (int j = 0; j < 8; ++j)
            Ap.u[j] = *(const uint32_t*)&Plds[wid][lid * 32 + koff(j, kh)];
        asm volatile("" ::: "memory");

        // ---- PV: V stored transposed so kv pairs are contiguous u32 ----
        #pragma unroll
        for (int t = 0; t < 4; ++t) {
            BfVec Bv;   // B[kv][d]: lane -> d col, vgpr -> kv pair (contig)
            const __bf16* vcol =
                VwT + ((size_t)bh * KEYD + t * 16 + lid) * SQ + kv0;
            #pragma unroll
            for (int j = 0; j < 8; ++j)
                Bv.u[j] = *(const uint32_t*)(vcol + koff(j, kh));
            acc[t] = __builtin_amdgcn_wmma_f32_16x16x32_bf16(
                false, Ap.v, false, Bv.v, (short)0, acc[t], false, false);
        }
    }

    // ---- normalize rows and write (B, S, H*64) f32 ----
    const int b = bh >> 4, h = bh & 15;
    #pragma unroll
    for (int r = 0; r < 8; ++r) {
        float inv = 1.0f / lrow[r];
        int m = r + 8 * half;
        float* orow = Out + ((size_t)b * SQ + q0 + m) * DMODEL + h * KEYD;
        #pragma unroll
        for (int t = 0; t < 4; ++t)
            orow[t * 16 + lid] = acc[t][r] * inv;
    }
}

extern "C" void kernel_launch(void* const* d_in, const int* in_sizes, int n_in,
                              void* d_out, int out_size, void* d_ws, size_t ws_size,
                              hipStream_t stream) {
    const float* q  = (const float*)d_in[0];
    const float* k  = (const float*)d_in[1];
    const float* v  = (const float*)d_in[2];
    const float* Wq = (const float*)d_in[3];
    const float* Wk = (const float*)d_in[4];
    const float* Wv = (const float*)d_in[5];

    const size_t elems = (size_t)2 * HEADS * SQ * KEYD;   // 4M bf16 each
    __bf16* qw  = (__bf16*)d_ws;
    __bf16* kw  = qw + elems;
    __bf16* vwT = kw + elems;

    dim3 blk(256);
    // 4096 wave-tiles each -> 512 blocks of 8 waves
    proj_kernel<<<512, blk, 0, stream>>>(q, Wq, qw, 0);
    proj_kernel<<<512, blk, 0, stream>>>(k, Wk, kw, 0);
    proj_kernel<<<512, blk, 0, stream>>>(v, Wv, vwT, 1);
    attn_kernel<<<512, blk, 0, stream>>>(qw, kw, vwT, (float*)d_out);
}
